// TriangleMultiplication_55851754717712
// MI455X (gfx1250) — matrix operand order repaired and checked by reading, not compiled, not run on hardware
//
#include <hip/hip_runtime.h>
#include <hip/hip_bf16.h>
#include <stdint.h>

// CDNA5 / gfx1250: wave32, WMMA bf16 16x16x32, f32 accumulate.

typedef __attribute__((ext_vector_type(16))) __bf16 v16bf;
typedef __attribute__((ext_vector_type(8)))  float  v8f;

#define LSEQ 512
#define DCH 128
#define NROWS (LSEQ * LSEQ)      // 262144 flattened positions
#define XS_STRIDE 136            // 128 + 8 bf16 pad (272B rows, 16B aligned, bank-spread)

static __device__ __forceinline__ uint16_t f2bf(float f) {
  union { float f; uint32_t u; } c; c.f = f;
  uint32_t r = c.u + 0x7FFFu + ((c.u >> 16) & 1u);   // round-to-nearest-even
  return (uint16_t)(r >> 16);
}
static __device__ __forceinline__ float bf2f(uint16_t h) {
  union { uint32_t u; float f; } c; c.u = ((uint32_t)h) << 16;
  return c.f;
}
static __device__ __forceinline__ float sigf(float x) {
  return 1.0f / (1.0f + __expf(-x));
}

union FragBF { v16bf v; uint4 q[2]; uint16_t h[16]; };

static __device__ __forceinline__ uint32_t pack2(float a, float b) {
  return (uint32_t)f2bf(a) | ((uint32_t)f2bf(b) << 16);
}

// ---------------------------------------------------------------------------
// Kernel 0: repack 6 weight matrices (k-major, 128x128 f32) into bf16 WMMA
// B-fragment order: [mat][ktile][ntile][lane][j], lane = (n&15)|((k>>4&1)<<4),
// j = k&15. A B fragment is then 32 contiguous bytes per lane.
// ---------------------------------------------------------------------------
__global__ void k_swizzle_w(const float* __restrict__ lpw, const float* __restrict__ lgw,
                            const float* __restrict__ rpw, const float* __restrict__ rgw,
                            const float* __restrict__ gw,  const float* __restrict__ opw,
                            uint16_t* __restrict__ wsW) {
  int gid = blockIdx.x * blockDim.x + threadIdx.x;
  if (gid >= 6 * DCH * DCH) return;
  int mat = gid / (DCH * DCH);
  int rem = gid % (DCH * DCH);
  int k = rem / DCH, n = rem % DCH;
  const float* src = (mat == 0) ? lpw : (mat == 1) ? lgw : (mat == 2) ? rpw
                   : (mat == 3) ? rgw : (mat == 4) ? gw  : opw;
  float v = src[k * DCH + n];
  int kt = k >> 5, nt = n >> 4;
  int lane = (n & 15) | (((k >> 4) & 1) << 4);
  int j = k & 15;
  wsW[((((mat << 2) + kt) << 3) + nt) * 512 + (lane << 4) + j] = f2bf(v);
}

// ---------------------------------------------------------------------------
// Kernel 1: fused LayerNorm(pair) + 5 projections (lp,lg,rp,rg,g) with fused
// sigmoid gating. Outputs channel-major bf16: leftT/rightT/gateT [128][NROWS].
// right is pre-scaled by 1/L. Block = 256 threads = 8 waves, 128 rows/block.
// ---------------------------------------------------------------------------
__global__ __launch_bounds__(256) void k_ln_proj(
    const float* __restrict__ pair,
    const float* __restrict__ nw, const float* __restrict__ nb,
    const float* __restrict__ lpb, const float* __restrict__ rpb,
    const float* __restrict__ lgb, const float* __restrict__ rgb,
    const float* __restrict__ gb,
    const uint16_t* __restrict__ wsW,
    uint16_t* __restrict__ leftT, uint16_t* __restrict__ rightT,
    uint16_t* __restrict__ gateT)
{
  __shared__ __align__(16) uint16_t xs[128 * XS_STRIDE];
  const int t = threadIdx.x;
  const int lane = t & 31;
  const int w = t >> 5;               // wave 0..7 -> 16-row M strip
  const int hi = lane >> 4;           // lane half
  const int nl = lane & 15;
  const size_t rowbase = (size_t)blockIdx.x * 128;

  // ---- Phase 1: LayerNorm, wave-per-row, bf16 tile into LDS ----
  float4 nwv = *(const float4*)(nw + lane * 4);
  float4 nbv = *(const float4*)(nb + lane * 4);
  for (int rr = 0; rr < 16; ++rr) {
    int row = w * 16 + rr;
    float4 v = *(const float4*)(pair + (rowbase + row) * DCH + lane * 4);
    float s  = v.x + v.y + v.z + v.w;
    float s2 = v.x * v.x + v.y * v.y + v.z * v.z + v.w * v.w;
    for (int off = 16; off >= 1; off >>= 1) {
      s  += __shfl_xor(s,  off);
      s2 += __shfl_xor(s2, off);
    }
    float m   = s * (1.0f / 128.0f);
    float var = s2 * (1.0f / 128.0f) - m * m;
    float rs  = rsqrtf(var + 1e-5f);
    uint2 pk;
    pk.x = pack2((v.x - m) * rs * nwv.x + nbv.x, (v.y - m) * rs * nwv.y + nbv.y);
    pk.y = pack2((v.z - m) * rs * nwv.z + nbv.z, (v.w - m) * rs * nwv.w + nbv.w);
    *(uint2*)&xs[row * XS_STRIDE + lane * 4] = pk;
  }
  __syncthreads();

  // ---- Phase 2: A fragments for this wave's 16-row strip (contiguous k = 16*hi + s, same as wsW B) ----
  FragBF af[4];
  {
    const uint16_t* base = &xs[(w * 16 + nl) * XS_STRIDE];
    for (int kt = 0; kt < 4; ++kt) {
      int c0 = kt * 32 + (hi ? 16 : 0);
      af[kt].q[0] = *(const uint4*)(base + c0);
      af[kt].q[1] = *(const uint4*)(base + c0 + 8);
    }
  }

  // GEMM over one 16x16 output tile: C = A(16x128) * W_mat(128, nt*16..)
  auto gemm_tile = [&](int mat, int nt) -> v8f {
    v8f acc = {};
    for (int kt = 0; kt < 4; ++kt) {
      FragBF bf;
      const uint16_t* p = wsW + ((((mat << 2) + kt) << 3) + nt) * 512 + lane * 16;
      bf.q[0] = *(const uint4*)p;
      bf.q[1] = *(const uint4*)(p + 8);
      acc = __builtin_amdgcn_wmma_f32_16x16x32_bf16(
          false, af[kt].v, false, bf.v, (short)0, acc, false, false);
    }
    return acc;
  };

  // Store 8 bf16 (rows m0..m0+7 of column nglob) as one 16B packed store.
  auto store_col = [&](uint16_t* dstT, int nglob, const float* o) {
    size_t grow = rowbase + (size_t)w * 16 + hi * 8;
    uint4 pk;
    pk.x = pack2(o[0], o[1]); pk.y = pack2(o[2], o[3]);
    pk.z = pack2(o[4], o[5]); pk.w = pack2(o[6], o[7]);
    *(uint4*)&dstT[(size_t)nglob * NROWS + grow] = pk;
  };

  for (int nt = 0; nt < 8; ++nt) {
    int nglob = nt * 16 + nl;
    float o[8];

    // left = sigmoid(x@lgw + lgb) * (x@lpw + lpb)
    v8f p = gemm_tile(0, nt);
    v8f g = gemm_tile(1, nt);
    float pbv = lpb[nglob], gbv = lgb[nglob];
    for (int r = 0; r < 8; ++r) o[r] = sigf(g[r] + gbv) * (p[r] + pbv);
    store_col(leftT, nglob, o);

    // right = sigmoid(x@rgw + rgb) * (x@rpw + rpb) * (1/L)
    p = gemm_tile(2, nt);
    g = gemm_tile(3, nt);
    pbv = rpb[nglob]; gbv = rgb[nglob];
    for (int r = 0; r < 8; ++r)
      o[r] = sigf(g[r] + gbv) * (p[r] + pbv) * (1.0f / (float)LSEQ);
    store_col(rightT, nglob, o);

    // gate = sigmoid(x@gw + gb)
    v8f gg = gemm_tile(4, nt);
    float gbb = gb[nglob];
    for (int r = 0; r < 8; ++r) o[r] = sigf(gg[r] + gbb);
    store_col(gateT, nglob, o);
  }
}

// ---------------------------------------------------------------------------
// Kernel 2: triangle einsum  tri[d][i][j] = sum_k leftT[d][i][k]*rightT[d][j][k]
// 128 batched 512x512x512 GEMMs (rightT's natural K-contiguous layout is the
// needed B^T). Block = 256x128 C tile of one d; 8 waves in a 4x2 grid, each
// wave owns a 64x64 patch: 16 f32 accumulators (128 VGPRs), 8 fragment loads
// per K-step for 16 WMMAs (1 b128-pair load per 2 WMMAs, ~87 FLOP/B from L2).
// ---------------------------------------------------------------------------
__global__ __launch_bounds__(256) void k_triangle(
    const uint16_t* __restrict__ leftT, const uint16_t* __restrict__ rightT,
    float* __restrict__ tri)
{
  const int t = threadIdx.x, lane = t & 31, w = t >> 5;
  const int hi = lane >> 4, nl = lane & 15;
  const int d = blockIdx.y;
  const int it = blockIdx.x >> 2, jt = blockIdx.x & 3;      // 2 x 4 block grid
  const int ibase = it * 256 + (w >> 1) * 64;               // wave M base (4 mtiles)
  const int jbase = jt * 128 + (w & 1) * 64;                // wave N base (4 ntiles)
  const uint16_t* A  = leftT  + (size_t)d * NROWS;
  const uint16_t* Bm = rightT + (size_t)d * NROWS;

  v8f acc[4][4];
  for (int mt = 0; mt < 4; ++mt)
    for (int nt = 0; nt < 4; ++nt) acc[mt][nt] = (v8f){};

  for (int k = 0; k < LSEQ; k += 32) {
    FragBF a[4], b[4];
    for (int mt = 0; mt < 4; ++mt) {
      const uint16_t* p = A + (size_t)(ibase + mt * 16 + nl) * LSEQ + k + (hi ? 16 : 0);
      a[mt].q[0] = *(const uint4*)p;
      a[mt].q[1] = *(const uint4*)(p + 8);
      __builtin_prefetch(p + 32, 0, 3);          // next K panel -> global_prefetch_b8
    }
    for (int nt = 0; nt < 4; ++nt) {
      const uint16_t* p = Bm + (size_t)(jbase + nt * 16 + nl) * LSEQ + k + (hi ? 16 : 0);
      b[nt].q[0] = *(const uint4*)p;
      b[nt].q[1] = *(const uint4*)(p + 8);
      __builtin_prefetch(p + 32, 0, 3);
    }
    for (int mt = 0; mt < 4; ++mt)
      for (int nt = 0; nt < 4; ++nt)
        acc[mt][nt] = __builtin_amdgcn_wmma_f32_16x16x32_bf16(
            false, a[mt].v, false, b[nt].v, (short)0, acc[mt][nt], false, false);
  }

  float* T = tri + (size_t)d * NROWS;
  for (int mt = 0; mt < 4; ++mt)
    for (int nt = 0; nt < 4; ++nt) {
      int n  = jbase + nt * 16 + nl;
      int m0 = ibase + mt * 16 + hi * 8;
      for (int r = 0; r < 8; ++r)
        T[(size_t)(m0 + r) * LSEQ + n] = acc[mt][nt][r];
    }
}

// ---------------------------------------------------------------------------
// Kernel 3: LayerNorm over d of tri (coalesced reads of [d][pos] layout),
// output projection (opw) via WMMA, + opb, * gate, store f32 to d_out.
// 128 positions per block; 8 waves each own one 16-row M strip x 8 N tiles.
// ---------------------------------------------------------------------------
__global__ __launch_bounds__(256) void k_ln_out(
    const float* __restrict__ tri,
    const float* __restrict__ noww, const float* __restrict__ nowb,
    const float* __restrict__ opb,
    const uint16_t* __restrict__ wsW,
    const uint16_t* __restrict__ gateT,
    float* __restrict__ out)
{
  __shared__ __align__(16) uint16_t xs[128 * XS_STRIDE];
  const int t = threadIdx.x, lane = t & 31, w = t >> 5;
  const int hi = lane >> 4, nl = lane & 15;
  const size_t pbase = (size_t)blockIdx.x * 128;

  // ---- Phase 1: LN over channels; two passes (tri tile is L2-hot) ----
  if (t < 128) {
    const float* src = tri + pbase + t;
    float s = 0.f, s2 = 0.f;
    for (int dd = 0; dd < DCH; ++dd) {
      float v = src[(size_t)dd * NROWS];
      s += v; s2 += v * v;
    }
    float m   = s * (1.0f / 128.0f);
    float var = s2 * (1.0f / 128.0f) - m * m;
    float rs  = rsqrtf(var + 1e-5f);
    for (int dd = 0; dd < DCH; ++dd) {
      float v = src[(size_t)dd * NROWS];
      xs[t * XS_STRIDE + dd] = f2bf((v - m) * rs * noww[dd] + nowb[dd]);
    }
  }
  __syncthreads();

  // ---- Phase 2: GEMM with opw (mat index 5 in wsW) ----
  FragBF af[4];
  {
    const uint16_t* base = &xs[(w * 16 + nl) * XS_STRIDE];
    for (int kt = 0; kt < 4; ++kt) {
      int c0 = kt * 32 + (hi ? 16 : 0);
      af[kt].q[0] = *(const uint4*)(base + c0);
      af[kt].q[1] = *(const uint4*)(base + c0 + 8);
    }
  }

  for (int nt = 0; nt < 8; ++nt) {
    v8f acc = {};
    for (int kt = 0; kt < 4; ++kt) {
      FragBF bf;
      const uint16_t* p = wsW + (((20 + kt) << 3) + nt) * 512 + lane * 16; // mat 5
      bf.q[0] = *(const uint4*)p;
      bf.q[1] = *(const uint4*)(p + 8);
      acc = __builtin_amdgcn_wmma_f32_16x16x32_bf16(
          false, af[kt].v, false, bf.v, (short)0, acc, false, false);
    }
    int nglob = nt * 16 + nl;
    float bias = opb[nglob];
    size_t m0 = pbase + (size_t)w * 16 + hi * 8;
    uint4 gq = *(const uint4*)&gateT[(size_t)nglob * NROWS + m0];  // 8 packed bf16 gates
    const uint16_t* gh = (const uint16_t*)&gq;
    for (int r = 0; r < 8; ++r)
      out[(m0 + r) * (size_t)DCH + nglob] = bf2f(gh[r]) * (acc[r] + bias);
  }
}

// ---------------------------------------------------------------------------
extern "C" void kernel_launch(void* const* d_in, const int* in_sizes, int n_in,
                              void* d_out, int out_size, void* d_ws, size_t ws_size,
                              hipStream_t stream) {
  (void)in_sizes; (void)n_in; (void)out_size; (void)ws_size;
  const float* pair   = (const float*)d_in[0];
  const float* norm_w = (const float*)d_in[1];
  const float* norm_b = (const float*)d_in[2];
  const float* lpw    = (const float*)d_in[3];
  const float* lpb    = (const float*)d_in[4];
  const float* rpw    = (const float*)d_in[5];
  const float* rpb    = (const float*)d_in[6];
  const float* lgw    = (const float*)d_in[7];
  const float* lgb    = (const float*)d_in[8];
  const float* rgw    = (const float*)d_in[9];
  const float* rgb    = (const float*)d_in[10];
  const float* gw     = (const float*)d_in[11];
  const float* gb     = (const float*)d_in[12];
  const float* now_w  = (const float*)d_in[13];
  const float* now_b  = (const float*)d_in[14];
  const float* opw    = (const float*)d_in[15];
  const float* opb    = (const float*)d_in[16];
  float* out = (float*)d_out;

  // Workspace layout (all 16B aligned):
  //   wsW    : 6 * 128*128 bf16 swizzled weights   (196,608 B)
  //   leftT  : [128][NROWS] bf16                   (64 MB)
  //   rightT : [128][NROWS] bf16 (pre-scaled 1/L)  (64 MB)
  //   gateT  : [128][NROWS] bf16                   (64 MB)
  //   tri    : [128][NROWS] f32                    (128 MB)
  char* ws = (char*)d_ws;
  uint16_t* wsW    = (uint16_t*)ws;
  uint16_t* leftT  = (uint16_t*)(ws + 196608);
  uint16_t* rightT = leftT  + (size_t)DCH * NROWS;
  uint16_t* gateT  = rightT + (size_t)DCH * NROWS;
  float*    tri    = (float*)(gateT + (size_t)DCH * NROWS);

  k_swizzle_w<<<(6 * DCH * DCH + 255) / 256, 256, 0, stream>>>(
      lpw, lgw, rpw, rgw, gw, opw, wsW);

  k_ln_proj<<<NROWS / 128, 256, 0, stream>>>(
      pair, norm_w, norm_b, lpb, rpb, lgb, rgb, gb, wsW, leftT, rightT, gateT);

  k_triangle<<<dim3(8, DCH), 256, 0, stream>>>(leftT, rightT, tri);

  k_ln_out<<<NROWS / 128, 256, 0, stream>>>(
      tri, now_w, now_b, opb, wsW, gateT, out);
}
